// FlowNet3DIMP_953482739750
// MI455X (gfx1250) — matrix operand order repaired and checked by reading, not compiled, not run on hardware
//
#include <hip/hip_runtime.h>
#include <hip/hip_bf16.h>
#include <stdint.h>

typedef __attribute__((ext_vector_type(16))) _Float16 v16h;
typedef __attribute__((ext_vector_type(8)))  _Float16 v8h;
typedef __attribute__((ext_vector_type(8)))  float    v8f;
typedef _Float16 half_t;

#define TPB 256
#define GRID1(total) dim3((unsigned)((((long)(total)) + TPB - 1) / TPB)), dim3(TPB), 0, stream

static inline int ceil_to(int x, int a) { return (x + a - 1) / a * a; }

// ---------------------------------------------------------------------------
// data movement / elementwise kernels
// ---------------------------------------------------------------------------

__global__ void transpose_bcn_kernel(const float* __restrict__ in, float* __restrict__ out,
                                     int C, int Npts, long total) {
  long t = (long)blockIdx.x * blockDim.x + threadIdx.x;
  if (t >= total) return;
  int n = (int)(t % Npts);
  long bc = t / Npts;
  int c = (int)(bc % C);
  int b = (int)(bc / C);
  out[((long)b * Npts + n) * C + c] = in[t];
}

// Convert f32 weight (K x N) into WMMA-B fragment order:
//   swz[(((tn * nkb + kb) * 32 + lane) * 16 + e]
//     = W[kb*32 + 8*(lane>>4) + (e&7) + 16*(e>>3)][tn*16 + (lane&15)]   (zero padded)
// so each lane's 16-element B fragment is one contiguous 32-byte load.
__global__ void convert_w_swz_kernel(const float* __restrict__ w, int K, int N,
                                     int Kpad, int Npad, half_t* __restrict__ out,
                                     long total) {
  long t = (long)blockIdx.x * blockDim.x + threadIdx.x;
  if (t >= total) return;
  int e = (int)(t & 15);
  int lane = (int)((t >> 4) & 31);
  long blk = t >> 9;                 // tn * nkb + kb
  int nkb = Kpad >> 5;
  int kb = (int)(blk % nkb);
  int tn = (int)(blk / nkb);
  int k = kb * 32 + 8 * (lane >> 4) + (e & 7) + 16 * (e >> 3);
  int n = tn * 16 + (lane & 15);
  float v = (k < K && n < N) ? w[(long)k * N + n] : 0.f;
  out[t] = (half_t)v;
}

__global__ void gather_rows_kernel(const float* __restrict__ in, const int* __restrict__ idx,
                                   int n, int s, int C, float* __restrict__ out, long total) {
  long t = (long)blockIdx.x * blockDim.x + threadIdx.x;
  if (t >= total) return;
  int c = (int)(t % C);
  long r = t / C;
  int si = (int)(r % s);
  int b = (int)(r / s);
  out[t] = in[((long)b * n + idx[(long)b * s + si]) * C + c];
}

// ---------------------------------------------------------------------------
// farthest point sampling: one workgroup per batch, dist array lives in LDS
// ---------------------------------------------------------------------------

__global__ void fps_kernel(const float* __restrict__ xyz, int n, int npoint,
                           int* __restrict__ out) {
  extern __shared__ unsigned char fps_smem[];
  float* dist = (float*)fps_smem;
  float* rv = dist + n;
  int* ri = (int*)(rv + 256);
  int b = blockIdx.x;
  const float* p = xyz + (size_t)b * n * 3;
  int t = threadIdx.x;
  for (int i = t; i < n; i += 256) dist[i] = 1.0e10f;
  __syncthreads();
  int far = 0;
  for (int s = 0; s < npoint; ++s) {
    if (t == 0) out[(long)b * npoint + s] = far;
    float fx = p[far * 3], fy = p[far * 3 + 1], fz = p[far * 3 + 2];
    float best = -1.f;
    int bi = n;
    for (int i = t; i < n; i += 256) {
      float dx = p[i * 3] - fx, dy = p[i * 3 + 1] - fy, dz = p[i * 3 + 2] - fz;
      float d = dx * dx + dy * dy + dz * dz;
      float dd = fminf(dist[i], d);
      dist[i] = dd;
      if (dd > best || (dd == best && i < bi)) { best = dd; bi = i; }
    }
    rv[t] = best; ri[t] = bi;
    __syncthreads();
    for (int o = 128; o > 0; o >>= 1) {
      if (t < o) {
        float ov = rv[t + o]; int oi = ri[t + o];
        if (ov > rv[t] || (ov == rv[t] && oi < ri[t])) { rv[t] = ov; ri[t] = oi; }
      }
      __syncthreads();
    }
    far = ri[0];
    __syncthreads();
  }
}

// ---------------------------------------------------------------------------
// ball query: wave32 per query; ballot-compacted first-nsample in index order
// ---------------------------------------------------------------------------

__global__ void ball_query_kernel(const float* __restrict__ xyz, const float* __restrict__ nxyz,
                                  int n, int s, float r2, int nsample,
                                  int* __restrict__ idx, long nwaves) {
  long wave = ((long)blockIdx.x * blockDim.x + threadIdx.x) >> 5;
  int lane = (int)(threadIdx.x & 31);
  if (wave >= nwaves) return;
  int si = (int)(wave % s);
  int b = (int)(wave / s);
  const float* p = xyz + (size_t)b * n * 3;
  const float* c = nxyz + ((size_t)b * s + si) * 3;
  float cx = c[0], cy = c[1], cz = c[2];
  int* o = idx + ((long)b * s + si) * nsample;
  int cnt = 0, first = -1;
  for (int base = 0; base < n; base += 32) {
    if (cnt >= nsample) break;
    int i = base + lane;
    bool pred = false;
    if (i < n) {
      float dx = p[i * 3] - cx, dy = p[i * 3 + 1] - cy, dz = p[i * 3 + 2] - cz;
      pred = (dx * dx + dy * dy + dz * dz <= r2);
    }
    unsigned mask = (unsigned)__ballot(pred);
    if (first < 0 && mask) first = base + (__ffs((int)mask) - 1);
    int pre = __popc(mask & ((1u << lane) - 1u));
    if (pred && (cnt + pre) < nsample) o[cnt + pre] = i;
    cnt += __popc(mask);
  }
  if (cnt > nsample) cnt = nsample;
  int fill = (first < 0) ? 0 : first;
  for (int j = cnt + lane; j < nsample; j += 32) o[j] = fill;
}

// ---------------------------------------------------------------------------
// kNN: thread per query, stable k-best insertion (matches top_k tie order)
// ---------------------------------------------------------------------------

template <int K>
__global__ void knn_kernel(const float* __restrict__ q, const float* __restrict__ ref,
                           int s, int n, int* __restrict__ oidx, float* __restrict__ od2,
                           long total) {
  long t = (long)blockIdx.x * blockDim.x + threadIdx.x;
  if (t >= total) return;
  int i = (int)(t % s);
  int b = (int)(t / s);
  const float* qp = q + ((long)b * s + i) * 3;
  float qx = qp[0], qy = qp[1], qz = qp[2];
  const float* rp = ref + (long)b * n * 3;
  float bd[K];
  int bi[K];
  for (int k = 0; k < K; ++k) { bd[k] = 3.0e38f; bi[k] = 0; }
  for (int j = 0; j < n; ++j) {
    float dx = rp[j * 3] - qx, dy = rp[j * 3 + 1] - qy, dz = rp[j * 3 + 2] - qz;
    float d = dx * dx + dy * dy + dz * dz;
    if (d < bd[K - 1]) {
      int pos = K - 1;
      while (pos > 0 && bd[pos - 1] > d) {
        bd[pos] = bd[pos - 1]; bi[pos] = bi[pos - 1]; --pos;
      }
      bd[pos] = d; bi[pos] = j;
    }
  }
  int* oo = oidx + t * K;
  for (int k = 0; k < K; ++k) oo[k] = bi[k];
  if (od2) {
    float* dd = od2 + t * K;
    for (int k = 0; k < K; ++k) dd[k] = bd[k];
  }
}

// ---------------------------------------------------------------------------
// grouping / concat kernels (write f16, zero-padded to K multiple of 32)
// ---------------------------------------------------------------------------

__global__ void group_sa_kernel(const float* __restrict__ xyz, const float* __restrict__ feat,
                                const float* __restrict__ nxyz, const int* __restrict__ idx,
                                int s, int ns, int n, int cf, int cpad,
                                half_t* __restrict__ out, long total) {
  long t = (long)blockIdx.x * blockDim.x + threadIdx.x;
  if (t >= total) return;
  int c = (int)(t % cpad);
  long row = t / cpad;
  int kj = (int)(row % ns);
  long bs = row / ns;
  int si = (int)(bs % s);
  int b = (int)(bs / s);
  int id = idx[((long)b * s + si) * ns + kj];
  float v = 0.f;
  if (c < 3) v = xyz[((long)b * n + id) * 3 + c] - nxyz[((long)b * s + si) * 3 + c];
  else if (c < 3 + cf) v = feat[((long)b * n + id) * cf + (c - 3)];
  out[t] = (half_t)v;
}

__global__ void group_fe_kernel(const float* __restrict__ x2, const float* __restrict__ f2,
                                const float* __restrict__ x1, const float* __restrict__ f1,
                                const int* __restrict__ idx, int s, int ns, int n,
                                int cf2, int cf1, int cpad, half_t* __restrict__ out, long total) {
  long t = (long)blockIdx.x * blockDim.x + threadIdx.x;
  if (t >= total) return;
  int c = (int)(t % cpad);
  long row = t / cpad;
  int kj = (int)(row % ns);
  long bs = row / ns;
  int si = (int)(bs % s);
  int b = (int)(bs / s);
  int id = idx[((long)b * s + si) * ns + kj];
  float v = 0.f;
  if (c < 3) v = x2[((long)b * n + id) * 3 + c] - x1[((long)b * s + si) * 3 + c];
  else if (c < 3 + cf2) v = f2[((long)b * n + id) * cf2 + (c - 3)];
  else if (c < 3 + cf2 + cf1) v = f1[((long)b * s + si) * cf1 + (c - 3 - cf2)];
  out[t] = (half_t)v;
}

__global__ void group_up_h_kernel(const float* __restrict__ x2, const float* __restrict__ f2,
                                  const float* __restrict__ x1, const int* __restrict__ idx,
                                  int s, int ns, int n, int cf2, int cpad,
                                  half_t* __restrict__ out, long total) {
  long t = (long)blockIdx.x * blockDim.x + threadIdx.x;
  if (t >= total) return;
  int c = (int)(t % cpad);
  long row = t / cpad;
  int kj = (int)(row % ns);
  long bs = row / ns;
  int si = (int)(bs % s);
  int b = (int)(bs / s);
  int id = idx[((long)b * s + si) * ns + kj];
  float v = 0.f;
  if (c < cf2) v = f2[((long)b * n + id) * cf2 + c];
  else if (c < cf2 + 3) v = x2[((long)b * n + id) * 3 + (c - cf2)] -
                            x1[((long)b * s + si) * 3 + (c - cf2)];
  out[t] = (half_t)v;
}

__global__ void group_up_f_kernel(const float* __restrict__ x2, const float* __restrict__ f2,
                                  const float* __restrict__ x1, const int* __restrict__ idx,
                                  int s, int ns, int n, int cf2, int cw,
                                  float* __restrict__ out, long total) {
  long t = (long)blockIdx.x * blockDim.x + threadIdx.x;
  if (t >= total) return;
  int c = (int)(t % cw);
  long row = t / cw;
  int kj = (int)(row % ns);
  long bs = row / ns;
  int si = (int)(bs % s);
  int b = (int)(bs / s);
  int id = idx[((long)b * s + si) * ns + kj];
  float v = 0.f;
  if (c < cf2) v = f2[((long)b * n + id) * cf2 + c];
  else v = x2[((long)b * n + id) * 3 + (c - cf2)] - x1[((long)b * s + si) * 3 + (c - cf2)];
  out[t] = v;
}

__global__ void concat_pad_h_kernel(const float* __restrict__ s0, int c0,
                                    const float* __restrict__ s1, int c1,
                                    const float* __restrict__ s2, int c2,
                                    int cpad, half_t* __restrict__ out, long total) {
  long t = (long)blockIdx.x * blockDim.x + threadIdx.x;
  if (t >= total) return;
  int c = (int)(t % cpad);
  long r = t / cpad;
  float v = 0.f;
  if (c < c0) v = s0[r * (size_t)c0 + c];
  else if (c < c0 + c1) v = s1[r * (size_t)c1 + (c - c0)];
  else if (c < c0 + c1 + c2) v = s2[r * (size_t)c2 + (c - c0 - c1)];
  out[t] = (half_t)v;
}

__global__ void interp_concat_kernel(const float* __restrict__ f2, const float* __restrict__ f1,
                                     const int* __restrict__ idx, const float* __restrict__ d2,
                                     int n1, int n2, int C2, int c1, int cpad,
                                     half_t* __restrict__ out, long total) {
  long t = (long)blockIdx.x * blockDim.x + threadIdx.x;
  if (t >= total) return;
  int c = (int)(t % cpad);
  long row = t / cpad;
  int i = (int)(row % n1);
  int b = (int)(row / n1);
  float v = 0.f;
  if (c < C2) {
    const float* dd = d2 + ((long)b * n1 + i) * 3;
    const int* ii = idx + ((long)b * n1 + i) * 3;
    float w0 = 1.f / fmaxf(dd[0], 1e-10f);
    float w1 = 1.f / fmaxf(dd[1], 1e-10f);
    float w2 = 1.f / fmaxf(dd[2], 1e-10f);
    float wsum = w0 + w1 + w2;
    w0 /= wsum; w1 /= wsum; w2 /= wsum;
    v = w0 * f2[((long)b * n2 + ii[0]) * C2 + c] +
        w1 * f2[((long)b * n2 + ii[1]) * C2 + c] +
        w2 * f2[((long)b * n2 + ii[2]) * C2 + c];
  } else if (c < C2 + c1) {
    v = f1[((long)b * n1 + i) * c1 + (c - C2)];
  }
  out[t] = (half_t)v;
}

// ---------------------------------------------------------------------------
// max-pool over nsample axis
// ---------------------------------------------------------------------------

__global__ void maxpool_h_kernel(const half_t* __restrict__ in, int ns, int ldi, int C,
                                 float* __restrict__ out, long total) {
  long t = (long)blockIdx.x * blockDim.x + threadIdx.x;
  if (t >= total) return;
  int c = (int)(t % C);
  long g = t / C;
  const half_t* p = in + (size_t)g * ns * ldi + c;
  float m = -3.0e38f;
  for (int k = 0; k < ns; ++k) m = fmaxf(m, (float)p[(size_t)k * ldi]);
  out[t] = m;
}

__global__ void maxpool_f_kernel(const float* __restrict__ in, int ns, int ldi, int C,
                                 float* __restrict__ out, long total) {
  long t = (long)blockIdx.x * blockDim.x + threadIdx.x;
  if (t >= total) return;
  int c = (int)(t % C);
  long g = t / C;
  const float* p = in + (size_t)g * ns * ldi + c;
  float m = -3.0e38f;
  for (int k = 0; k < ns; ++k) m = fmaxf(m, p[(size_t)k * ldi]);
  out[t] = m;
}

// ---------------------------------------------------------------------------
// WMMA GEMM: one 16x16 f32 tile per wave, K in steps of 32 (f16 inputs).
// A: two contiguous v8h (16B) loads per lane per K-step (ISA 16-bit A layout).
// B: pre-swizzled to per-lane fragment order -> one contiguous v16h load.
// EXEC is all-ones around every v_wmma (wave-uniform guards only).
// ---------------------------------------------------------------------------

__global__ void wmma_gemm_kernel(const half_t* __restrict__ A, const half_t* __restrict__ Wswz,
                                 float* __restrict__ Z, const float* __restrict__ bias,
                                 int M, int NpadW, int Kpad, int lda, int Nout, int ldz) {
  int ntn = NpadW >> 4;
  int nkb = Kpad >> 5;
  long tiles = (long)(M >> 4) * ntn;
  long wave = (long)blockIdx.x * (blockDim.x >> 5) + (threadIdx.x >> 5);
  if (wave >= tiles) return;
  int tm = (int)(wave / ntn);
  int tn = (int)(wave % ntn);
  int lane = (int)(threadIdx.x & 31);
  int mrow = lane & 15;
  int h = lane >> 4;
  const half_t* arow = A + ((size_t)tm * 16 + mrow) * (size_t)lda + 8 * h;
  const v16h* wp = (const v16h*)Wswz + (size_t)tn * nkb * 32 + lane;
  int ncol = tn * 16 + mrow;
  v8f acc = {};
  for (int kb = 0; kb < nkb; ++kb) {
    const half_t* ap = arow + (kb << 5);
    v8h alo = *(const v8h*)(ap);
    v8h ahi = *(const v8h*)(ap + 16);
    v16h bv = wp[(size_t)kb * 32];
    if (kb + 1 < nkb) {
      __builtin_prefetch(ap + 32, 0, 1);
      __builtin_prefetch((const void*)(wp + (size_t)(kb + 1) * 32), 0, 1);
    }
    v16h av;
#pragma unroll
    for (int e = 0; e < 8; ++e) {
      av[e] = alo[e];
      av[8 + e] = ahi[e];
    }
    acc = __builtin_amdgcn_wmma_f32_16x16x32_f16(false, av, false, bv, (short)0, acc,
                                                 false, false);
  }
  if (ncol < Nout) {
#pragma unroll
    for (int r = 0; r < 8; ++r) {
      float v = acc[r];
      if (bias) v += bias[ncol];
      Z[((size_t)tm * 16 + r + 8 * h) * (size_t)ldz + ncol] = v;
    }
  }
}

// deterministic per-channel sum / sumsq (one block per channel)
__global__ void col_stats_kernel(const float* __restrict__ Z, int M, int ldz,
                                 float* __restrict__ stats, int sstride) {
  int c = blockIdx.x;
  int t = threadIdx.x;
  float s1 = 0.f, s2 = 0.f;
  for (int r = t; r < M; r += 256) {
    float v = Z[(size_t)r * ldz + c];
    s1 += v;
    s2 += v * v;
  }
  __shared__ float a1[256], a2[256];
  a1[t] = s1; a2[t] = s2;
  __syncthreads();
  for (int o = 128; o > 0; o >>= 1) {
    if (t < o) { a1[t] += a1[t + o]; a2[t] += a2[t + o]; }
    __syncthreads();
  }
  if (t == 0) { stats[c] = a1[0]; stats[sstride + c] = a2[0]; }
}

__global__ void bn_relu_kernel(const float* __restrict__ Z, const float* __restrict__ stats,
                               int M, int N, int ldz, int sstride,
                               half_t* __restrict__ outH, int ldo,
                               float* __restrict__ outF, long total) {
  long t = (long)blockIdx.x * blockDim.x + threadIdx.x;
  if (t >= total) return;
  int c = (int)(t % ldo);
  long row = t / ldo;
  float y = 0.f;
  if (c < N) {
    float inv = 1.f / (float)M;
    float mean = stats[c] * inv;
    float var = stats[sstride + c] * inv - mean * mean;
    float x = Z[row * (size_t)ldz + c];
    y = fmaxf((x - mean) * rsqrtf(var + 1e-5f), 0.f);
    if (outF) outF[row * (size_t)N + c] = y;
  }
  if (outH) outH[t] = (half_t)y;
}

// ---------------------------------------------------------------------------
// host orchestration
// ---------------------------------------------------------------------------

struct HW {
  const half_t* w;
  int Kpad, NpadW, N, ldo;
};

extern "C" void kernel_launch(void* const* d_in, const int* in_sizes, int n_in,
                              void* d_out, int out_size, void* d_ws, size_t ws_size,
                              hipStream_t stream) {
  (void)in_sizes; (void)out_size; (void)ws_size;
  if (n_in < 37 || !d_out || !d_ws) return;
  const int B = 2, N0 = 8192, N1 = 4096, N2 = 1024, N3 = 256, N4 = 64;

  char* base = (char*)d_ws;
  size_t off = 0;
  auto take = [&](size_t elems, size_t esz) -> void* {
    off = (off + 255) & ~(size_t)255;
    void* p = base + off;
    off += elems * esz;
    return p;
  };
  auto takef = [&](size_t e) { return (float*)take(e, 4); };
  auto takei = [&](size_t e) { return (int*)take(e, 4); };
  auto takeh = [&](size_t e) { return (half_t*)take(e, 2); };

  // persistent point / feature buffers (f32)
  float* x1f = takef((size_t)B * N0 * 3);
  float* x2f = takef((size_t)B * N0 * 3);
  float* f1f = takef((size_t)B * N0 * 3);
  float* f2f = takef((size_t)B * N0 * 3);
  float* l0f1 = takef((size_t)B * N0 * 24);
  float* l0f2 = takef((size_t)B * N0 * 24);
  float* l1x1 = takef((size_t)B * N1 * 3);
  float* l1x2 = takef((size_t)B * N1 * 3);
  float* l1f1 = takef((size_t)B * N1 * 64);
  float* l1f2 = takef((size_t)B * N1 * 64);
  float* l2x1 = takef((size_t)B * N2 * 3);
  float* l2x2 = takef((size_t)B * N2 * 3);
  float* l2f1 = takef((size_t)B * N2 * 128);
  float* l2f2 = takef((size_t)B * N2 * 128);
  float* l2f1n = takef((size_t)B * N2 * 128);
  float* l3x1 = takef((size_t)B * N3 * 3);
  float* l3f1 = takef((size_t)B * N3 * 256);
  float* l4x1 = takef((size_t)B * N4 * 3);
  float* l4f1 = takef((size_t)B * N4 * 512);
  float* l3fn = takef((size_t)B * N3 * 256);
  float* l2fn = takef((size_t)B * N2 * 256);
  float* l1fn = takef((size_t)B * N1 * 256);
  float* POOL = takef((size_t)B * N1 * 256);
  int* fpsidx = takei((size_t)B * N1);
  int* ballidx = takei((size_t)B * N0 * 16);
  int* knnidx = takei((size_t)B * N2 * 64);
  float* knnd2 = takef((size_t)B * N0 * 3);
  float* STATS = takef(1024);

  // f16 weights in WMMA fragment order; K->mult32, N->mult16 zero-padded
  auto mkw = [&](int idx, int K, int N_) {
    HW h;
    h.Kpad = ceil_to(K, 32);
    h.NpadW = ceil_to(N_, 16);
    h.N = N_;
    h.ldo = ceil_to(N_, 32);
    long tot = (long)h.Kpad * h.NpadW;  // == (NpadW/16)*(Kpad/32)*512
    half_t* dst = takeh((size_t)tot);
    convert_w_swz_kernel<<<GRID1(tot)>>>((const float*)d_in[idx], K, N_, h.Kpad, h.NpadW,
                                         dst, tot);
    h.w = dst;
    return h;
  };
  // params flattened in dict-insertion order (su1_m1 is empty)
  HW sa0w[3] = {mkw(4, 6, 16), mkw(5, 16, 16), mkw(6, 16, 24)};
  HW sa1w[3] = {mkw(7, 27, 64), mkw(8, 64, 64), mkw(9, 64, 64)};
  HW sa2w[3] = {mkw(10, 67, 64), mkw(11, 64, 64), mkw(12, 64, 128)};
  HW sa3w[3] = {mkw(13, 131, 128), mkw(14, 128, 128), mkw(15, 128, 256)};
  HW sa4w[3] = {mkw(16, 259, 256), mkw(17, 256, 256), mkw(18, 256, 512)};
  HW few[3]  = {mkw(19, 259, 128), mkw(20, 128, 128), mkw(21, 128, 128)};
  HW su1m2[2] = {mkw(22, 771, 256), mkw(23, 256, 256)};
  HW su2m1[3] = {mkw(24, 259, 128), mkw(25, 128, 128), mkw(26, 128, 256)};
  HW su2m2[1] = {mkw(27, 512, 256)};
  HW su3m1[3] = {mkw(28, 259, 128), mkw(29, 128, 128), mkw(30, 128, 256)};
  HW su3m2[1] = {mkw(31, 320, 256)};
  HW fpw[2]  = {mkw(32, 259, 256), mkw(33, 256, 256)};
  HW c1w = mkw(34, 256, 128);
  HW c2w = mkw(35, 128, 3);
  const float* conv2b = (const float*)d_in[36];

  // big reusable arenas
  half_t* HBUF0 = takeh((size_t)B * N2 * 64 * 288);  // 37.7M f16
  half_t* HBUF1 = takeh((size_t)B * N2 * 64 * 128);  // 16.8M f16
  float* ZBUF = takef((size_t)B * N2 * 64 * 128);    // 16.8M f32

  auto run_layer = [&](const half_t* Ain, long rows, int lda, const HW& w,
                       half_t* outH, float* outF) {
    long tiles = (rows >> 4) * (long)(w.NpadW >> 4);
    wmma_gemm_kernel<<<dim3((unsigned)((tiles + 3) / 4)), dim3(128), 0, stream>>>(
        Ain, w.w, ZBUF, nullptr, (int)rows, w.NpadW, w.Kpad, lda, w.N, w.NpadW);
    col_stats_kernel<<<dim3((unsigned)w.N), dim3(256), 0, stream>>>(
        ZBUF, (int)rows, w.NpadW, STATS, w.NpadW);
    long total = rows * w.ldo;
    bn_relu_kernel<<<GRID1(total)>>>(ZBUF, STATS, (int)rows, w.N, w.NpadW, w.NpadW,
                                     outH, w.ldo, outF, total);
  };

  auto run_mlp = [&](long rows, int cpad0, const HW* ws, int nl, float* outF,
                     half_t** lastH, int* lastLd) {
    const half_t* cur = HBUF0;
    int lda = cpad0;
    half_t* oH = HBUF1;
    for (int i = 0; i < nl; ++i) {
      oH = (cur == HBUF0) ? HBUF1 : HBUF0;
      run_layer(cur, rows, lda, ws[i], oH, (i == nl - 1) ? outF : nullptr);
      cur = oH;
      lda = ws[i].ldo;
    }
    if (lastH) *lastH = oH;
    if (lastLd) *lastLd = lda;
  };

  auto run_sa = [&](const float* x, const float* feat, int n, int cf, int npoint,
                    float radius, int ns, const HW* ws, int nl,
                    float* out_x, float* out_f, bool group_all) {
    const float* nx;
    int s;
    if (group_all) { nx = x; s = n; }
    else {
      s = npoint;
      size_t lds = (size_t)n * 4 + 256 * 8;
      fps_kernel<<<dim3(B), dim3(256), lds, stream>>>(x, n, npoint, fpsidx);
      long gt = (long)B * s * 3;
      gather_rows_kernel<<<GRID1(gt)>>>(x, fpsidx, n, s, 3, out_x, gt);
      nx = out_x;
    }
    long nw = (long)B * s;
    ball_query_kernel<<<GRID1(nw * 32)>>>(x, nx, n, s, radius * radius, ns, ballidx, nw);
    int cpad = ceil_to(3 + cf, 32);
    long rows = (long)B * s * ns;
    long tot = rows * cpad;
    group_sa_kernel<<<GRID1(tot)>>>(x, feat, nx, ballidx, s, ns, n, cf, cpad, HBUF0, tot);
    half_t* lastH;
    int lastLd;
    run_mlp(rows, cpad, ws, nl, nullptr, &lastH, &lastLd);
    int C = ws[nl - 1].N;
    long pt = (long)B * s * C;
    maxpool_h_kernel<<<GRID1(pt)>>>(lastH, ns, lastLd, C, out_f, pt);
  };

  auto run_upconv = [&](const float* x1p, const float* x2p, int s, int n,
                        const float* f2, int cf2, const float* f1a, int c1a,
                        const float* f1b, int c1b, const HW* w1, int nl1,
                        const HW* w2, int nl2, float* outF) {
    knn_kernel<8><<<GRID1((long)B * s)>>>(x1p, x2p, s, n, knnidx, nullptr, (long)B * s);
    long rows = (long)B * s * 8;
    int pooledC;
    if (nl1 > 0) {
      int cpad = ceil_to(cf2 + 3, 32);
      long tot = rows * cpad;
      group_up_h_kernel<<<GRID1(tot)>>>(x2p, f2, x1p, knnidx, s, 8, n, cf2, cpad, HBUF0, tot);
      half_t* lastH;
      int lastLd;
      run_mlp(rows, cpad, w1, nl1, nullptr, &lastH, &lastLd);
      pooledC = w1[nl1 - 1].N;
      long pt = (long)B * s * pooledC;
      maxpool_h_kernel<<<GRID1(pt)>>>(lastH, 8, lastLd, pooledC, POOL, pt);
    } else {
      pooledC = cf2 + 3;
      long tot = rows * pooledC;
      group_up_f_kernel<<<GRID1(tot)>>>(x2p, f2, x1p, knnidx, s, 8, n, cf2, pooledC, ZBUF, tot);
      long pt = (long)B * s * pooledC;
      maxpool_f_kernel<<<GRID1(pt)>>>(ZBUF, 8, pooledC, pooledC, POOL, pt);
    }
    int ctot = pooledC + c1a + c1b;
    int cpad2 = ceil_to(ctot, 32);
    long rows2 = (long)B * s;
    long tot2 = rows2 * cpad2;
    concat_pad_h_kernel<<<GRID1(tot2)>>>(POOL, pooledC, f1a, c1a, f1b, c1b, cpad2, HBUF0, tot2);
    half_t* lastH;
    int lastLd;
    run_mlp(rows2, cpad2, w2, nl2, outF, &lastH, &lastLd);
  };

  // ---- forward pass ----
  long tt = (long)B * 3 * N0;
  transpose_bcn_kernel<<<GRID1(tt)>>>((const float*)d_in[0], x1f, 3, N0, tt);
  transpose_bcn_kernel<<<GRID1(tt)>>>((const float*)d_in[1], x2f, 3, N0, tt);
  transpose_bcn_kernel<<<GRID1(tt)>>>((const float*)d_in[2], f1f, 3, N0, tt);
  transpose_bcn_kernel<<<GRID1(tt)>>>((const float*)d_in[3], f2f, 3, N0, tt);

  run_sa(x1f, f1f, N0, 3, 0, 0.2f, 16, sa0w, 3, nullptr, l0f1, true);
  run_sa(x2f, f2f, N0, 3, 0, 0.2f, 16, sa0w, 3, nullptr, l0f2, true);
  run_sa(x1f, l0f1, N0, 24, N1, 0.2f, 16, sa1w, 3, l1x1, l1f1, false);
  run_sa(x2f, l0f2, N0, 24, N1, 0.2f, 16, sa1w, 3, l1x2, l1f2, false);
  run_sa(l1x1, l1f1, N1, 64, N2, 0.4f, 16, sa2w, 3, l2x1, l2f1, false);
  run_sa(l1x2, l1f2, N1, 64, N2, 0.4f, 16, sa2w, 3, l2x2, l2f2, false);

  // flow embedding (k=64)
  {
    knn_kernel<64><<<GRID1((long)B * N2)>>>(l2x1, l2x2, N2, N2, knnidx, nullptr, (long)B * N2);
    long rows = (long)B * N2 * 64;
    int cpad = ceil_to(3 + 128 + 128, 32);  // 288
    long tot = rows * cpad;
    group_fe_kernel<<<GRID1(tot)>>>(l2x2, l2f2, l2x1, l2f1, knnidx, N2, 64, N2, 128, 128,
                                    cpad, HBUF0, tot);
    half_t* lastH;
    int lastLd;
    run_mlp(rows, cpad, few, 3, nullptr, &lastH, &lastLd);
    long pt = (long)B * N2 * 128;
    maxpool_h_kernel<<<GRID1(pt)>>>(lastH, 64, lastLd, 128, l2f1n, pt);
  }

  run_sa(l2x1, l2f1n, N2, 128, N3, 0.8f, 8, sa3w, 3, l3x1, l3f1, false);
  run_sa(l3x1, l3f1, N3, 256, N4, 1.6f, 8, sa4w, 3, l4x1, l4f1, false);

  run_upconv(l3x1, l4x1, N3, N4, l4f1, 512, l3f1, 256, nullptr, 0, nullptr, 0, su1m2, 2, l3fn);
  run_upconv(l2x1, l3x1, N2, N3, l3fn, 256, l2f1, 128, l2f1n, 128, su2m1, 3, su2m2, 1, l2fn);
  run_upconv(l1x1, l2x1, N1, N2, l2fn, 256, l1f1, 64, nullptr, 0, su3m1, 3, su3m2, 1, l1fn);

  // feature propagation + final conv
  {
    knn_kernel<3><<<GRID1((long)B * N0)>>>(x1f, l1x1, N0, N1, knnidx, knnd2, (long)B * N0);
    long rows = (long)B * N0;
    int cpad = ceil_to(256 + 3, 32);  // 288
    long tot = rows * cpad;
    interp_concat_kernel<<<GRID1(tot)>>>(l1fn, f1f, knnidx, knnd2, N0, N1, 256, 3, cpad,
                                         HBUF0, tot);
    half_t* lastH;
    int lastLd;
    run_mlp(rows, cpad, fpw, 2, nullptr, &lastH, &lastLd);
    half_t* convH = (lastH == HBUF0) ? HBUF1 : HBUF0;
    run_layer(lastH, rows, lastLd, c1w, convH, nullptr);
    long tiles = (rows >> 4) * (long)(c2w.NpadW >> 4);
    wmma_gemm_kernel<<<dim3((unsigned)((tiles + 3) / 4)), dim3(128), 0, stream>>>(
        convH, c2w.w, (float*)d_out, conv2b, (int)rows, c2w.NpadW, c2w.Kpad, c1w.ldo, 3, 3);
  }
}
